// MixerMamba2_23965917512244
// MI455X (gfx1250) — matrix operand order repaired and checked by reading, not compiled, not run on hardware
//
#include <hip/hip_runtime.h>
#include <hip/hip_bf16.h>

typedef __attribute__((ext_vector_type(16))) __bf16 v16bf;
typedef __attribute__((ext_vector_type(2)))  __bf16 v2bf;
typedef __attribute__((ext_vector_type(8)))  float  v8f;

// ---------------- constants ----------------
#define DMODEL 1024
#define DINNER 2048
#define DSTATE 128
#define NHEADS 32
#define HEADDIM 64
#define DCONV 4
#define CHUNK 256
#define NBATCH 2
#define SEQLEN 2048
#define BL (NBATCH*SEQLEN)            // 4096 tokens
#define DPROJ 4384                    // 2*2048 + 2*128 + 32
#define CONVDIM 2304                  // 2048 + 2*128
#define NCHUNK (SEQLEN/CHUNK)         // 8
#define NBCH (NBATCH*NCHUNK*NHEADS)   // 512

// ---------------- helpers ----------------
static __device__ __forceinline__ unsigned short f32_bf16r(float f) {
  unsigned u = __builtin_bit_cast(unsigned, f);
  unsigned r = u + 0x7FFFu + ((u >> 16) & 1u);
  return (unsigned short)(r >> 16);
}
static __device__ __forceinline__ float bf16_f32(unsigned short h) {
  unsigned u = ((unsigned)h) << 16;
  return __builtin_bit_cast(float, u);
}

static __device__ __forceinline__ v8f zero8() {
  v8f z;
  #pragma unroll
  for (int i = 0; i < 8; ++i) z[i] = 0.f;
  return z;
}

// Async 16B copy global -> LDS (GLOBAL_LOAD_ASYNC_TO_LDS_B128, tracked by ASYNCcnt).
// LDS byte address = low 32 bits of the generic pointer (LDS aperture encodes the
// offset in addr[31:0], ISA 10.2).
static __device__ __forceinline__ void async_copy16(const unsigned short* gsrc,
                                                    unsigned short* ldst) {
  unsigned lds_off = (unsigned)(unsigned long long)(const void*)ldst;
  unsigned long long gaddr = (unsigned long long)(const void*)gsrc;
  asm volatile("global_load_async_to_lds_b128 %0, %1, off"
               :: "v"(lds_off), "v"(gaddr) : "memory");
}
static __device__ __forceinline__ void async_wait() {
  asm volatile("s_wait_asynccnt 0x0" ::: "memory");
}

// A fragment (16x32 bf16): lds layout [row][k], k contiguous, stride ldk (ushort units).
static __device__ __forceinline__ v16bf frag_a(const unsigned short* lds, int row0,
                                               int ldk, int k0, int lane) {
  int r  = row0 + (lane & 15);
  int kk = (lane >> 4) << 3;       // 0 or 8
  const unsigned short* p = lds + r * ldk + k0 + kk;
  v16bf a;
  #pragma unroll
  for (int i = 0; i < 4; ++i) {
    unsigned u0 = *(const unsigned*)(p + 2 * i);          // K = kk+2i, kk+2i+1
    unsigned u1 = *(const unsigned*)(p + 16 + 2 * i);     // K = 16+kk+2i, ...
    v2bf x0 = __builtin_bit_cast(v2bf, u0);
    v2bf x1 = __builtin_bit_cast(v2bf, u1);
    a[2 * i]     = x0[0]; a[2 * i + 1]     = x0[1];
    a[8 + 2 * i] = x1[0]; a[8 + 2 * i + 1] = x1[1];
  }
  return a;
}

// B fragment (32x16 bf16): lds layout [col][k] (B^T, k contiguous), stride ldk.
static __device__ __forceinline__ v16bf frag_b(const unsigned short* lds, int col0,
                                               int ldk, int k0, int lane) {
  int n  = col0 + (lane & 15);
  int kb = (lane >> 4) << 3;       // 0 or 8
  const unsigned short* p = lds + n * ldk + k0 + kb;
  v16bf b;
  #pragma unroll
  for (int i = 0; i < 8; ++i) {
    unsigned u = *(const unsigned*)(p + 2 * i + (i >> 2) * 8);
    v2bf x = __builtin_bit_cast(v2bf, u);
    b[2 * i] = x[0]; b[2 * i + 1] = x[1];
  }
  return b;
}

static __device__ __forceinline__ v8f wmma_bf16(v16bf a, v16bf b, v8f c) {
  return __builtin_amdgcn_wmma_f32_16x16x32_bf16(false, a, false, b, (short)0, c,
                                                 false, false);
}

static __device__ __forceinline__ float silu(float x) {
  return x / (1.f + __expf(-x));
}

// ================= K0: f32 -> bf16 bulk convert =================
__global__ __launch_bounds__(256) void k_cvt(const float* __restrict__ src,
                                             unsigned short* __restrict__ dst, long n) {
  long i = (long)blockIdx.x * 256 + threadIdx.x;
  if (i < n) dst[i] = f32_bf16r(src[i]);
}

// ================= K1: zxbcdt = u @ W_in^T  (4096 x 4384, K=1024) =================
#define BLK_M 128
#define BLK_N 128
#define BLK_K 32
__global__ __launch_bounds__(256) void k_in_proj(const unsigned short* __restrict__ Ub,
                                                 const unsigned short* __restrict__ Wb,
                                                 float* __restrict__ ZX) {
  __shared__ __align__(16) unsigned short As[BLK_M][BLK_K];   // u tile  [m][k]
  __shared__ __align__(16) unsigned short Bs[BLK_N][BLK_K];   // W tile  [n][k]
  const int tid  = threadIdx.x;
  const int lane = tid & 31;
  const int wave = tid >> 5;              // 0..7
  const int wm   = wave >> 1;             // 0..3 : 32-row band
  const int wn   = wave & 1;              // 0..1 : 64-col band
  const int m_base = blockIdx.y * BLK_M;
  const int n_base = blockIdx.x * BLK_N;

  v8f acc[2][4];
  #pragma unroll
  for (int i = 0; i < 2; ++i)
    #pragma unroll
    for (int j = 0; j < 4; ++j) acc[i][j] = zero8();

  for (int k0 = 0; k0 < DMODEL; k0 += BLK_K) {
    // stage A: 128 rows x 32 k = 512 x 16B units, async copies
    #pragma unroll
    for (int t = tid; t < (BLK_M * BLK_K) / 8; t += 256) {
      int m = t >> 2, seg = t & 3;
      async_copy16(&Ub[(size_t)(m_base + m) * DMODEL + k0 + seg * 8], &As[m][seg * 8]);
    }
    #pragma unroll
    for (int t = tid; t < (BLK_N * BLK_K) / 8; t += 256) {
      int n = t >> 2, seg = t & 3;
      int gn = n_base + n;
      unsigned short* dst = &Bs[n][seg * 8];
      if (gn < DPROJ) {
        async_copy16(&Wb[(size_t)gn * DMODEL + k0 + seg * 8], dst);
      } else {
        unsigned* d = (unsigned*)dst;
        d[0] = 0; d[1] = 0; d[2] = 0; d[3] = 0;
      }
    }
    async_wait();
    __syncthreads();
    #pragma unroll
    for (int i = 0; i < 2; ++i) {
      v16bf af = frag_a(&As[0][0], wm * 32 + i * 16, BLK_K, 0, lane);
      #pragma unroll
      for (int j = 0; j < 4; ++j) {
        v16bf bf = frag_b(&Bs[0][0], wn * 64 + j * 16, BLK_K, 0, lane);
        acc[i][j] = wmma_bf16(af, bf, acc[i][j]);
      }
    }
    __syncthreads();
  }
  const int row_hi = (lane >> 4) ? 8 : 0;
  const int colL   = lane & 15;
  #pragma unroll
  for (int i = 0; i < 2; ++i)
    #pragma unroll
    for (int j = 0; j < 4; ++j) {
      int gm0 = m_base + wm * 32 + i * 16 + row_hi;
      int gn  = n_base + wn * 64 + j * 16 + colL;
      if (gn < DPROJ) {
        #pragma unroll
        for (int r = 0; r < 8; ++r)
          ZX[(size_t)(gm0 + r) * DPROJ + gn] = acc[i][j][r];
      }
    }
}

// ================= K2: causal depthwise conv1d + SiLU (bf16 out) =================
__global__ __launch_bounds__(256) void k_conv(const float* __restrict__ ZX,
                                              const float* __restrict__ cw,
                                              const float* __restrict__ cb,
                                              unsigned short* __restrict__ XCb) {
  long idx = (long)blockIdx.x * 256 + threadIdx.x;   // over BL*CONVDIM
  if (idx >= (long)BL * CONVDIM) return;
  int  c  = (int)(idx % CONVDIM);
  long bl = idx / CONVDIM;
  int  l  = (int)(bl % SEQLEN);
  float acc = cb[c];
  #pragma unroll
  for (int k = 0; k < DCONV; ++k) {
    int ls = l - (DCONV - 1) + k;
    float v = (ls >= 0) ? ZX[(bl + (ls - l)) * DPROJ + DINNER + c] : 0.f;
    acc += v * cw[c * DCONV + k];
  }
  XCb[idx] = f32_bf16r(silu(acc));
}

// ================= K2b: dt = softplus(dt_raw + dt_bias) =================
__global__ __launch_bounds__(256) void k_dt(const float* __restrict__ ZX,
                                            const float* __restrict__ dt_bias,
                                            float* __restrict__ DT) {
  int idx = blockIdx.x * 256 + threadIdx.x;          // over BL*NHEADS
  if (idx >= BL * NHEADS) return;
  int  h  = idx & (NHEADS - 1);
  long bl = idx >> 5;
  float x = ZX[bl * DPROJ + (DPROJ - NHEADS) + h] + dt_bias[h];
  DT[idx] = (x > 20.f) ? x : __logf(1.f + __expf(x));
}

// ================= K3: per-(b,c,h) inclusive cumsum of a = dt*A =================
__global__ __launch_bounds__(256) void k_cumsum(const float* __restrict__ DT,
                                                const float* __restrict__ A_log,
                                                float* __restrict__ ACUM) {
  __shared__ float s[CHUNK];
  int h  = blockIdx.x & 31;
  int bc = blockIdx.x >> 5;      // b*8 + c
  int b  = bc >> 3, c = bc & 7;
  int l  = threadIdx.x;
  long bl = (long)b * SEQLEN + c * CHUNK + l;
  float A = -__expf(A_log[h]);
  s[l] = DT[bl * NHEADS + h] * A;
  __syncthreads();
  for (int off = 1; off < CHUNK; off <<= 1) {
    float v = (l >= off) ? s[l - off] : 0.f;
    __syncthreads();
    s[l] += v;
    __syncthreads();
  }
  ACUM[(long)blockIdx.x * CHUNK + l] = s[l];
}

// ================= K4: intra-chunk Y_diag + chunk states (per b,c,h) =============
__global__ __launch_bounds__(256) void k_chunk(const unsigned short* __restrict__ XCb,
                                               const float* __restrict__ DT,
                                               const float* __restrict__ ACUM,
                                               float* __restrict__ Y,
                                               float* __restrict__ STATES) {
  __shared__ __align__(16) unsigned short Cr[64][128];   // C rows block [i][n]
  __shared__ __align__(16) unsigned short Bsl[64][128];  // B rows block [j][n]
  __shared__ __align__(16) unsigned short XT[64][64];    // x^T [p][j]
  __shared__ __align__(16) unsigned short Gs[64][64];    // masked G*dt bf16 [i][j]
  __shared__ float acR[64], acS[64], dtS[64];

  const int bid  = blockIdx.x;               // (b*8+c)*32 + h
  const int h    = bid & 31;
  const int bc   = bid >> 5;
  const int b    = bc >> 3, c = bc & 7;
  const int tid  = threadIdx.x;
  const int lane = tid & 31;
  const int wave = tid >> 5;
  const int row_hi = (lane >> 4) ? 8 : 0;
  const int colL   = lane & 15;
  const long tok0  = (long)b * SEQLEN + c * CHUNK;
  const long acb   = (long)bid * CHUNK;

  const int m0 = (wave >> 1) * 16;           // output row band (16)
  const int nb = (wave & 1) * 32;            // output col band (32)

  // ---------- phase 1: Y_diag ----------
  for (int R = 0; R < 4; ++R) {
    // async-stage C rows: 64 x 128 bf16 = 1024 x 16B
    for (int t = tid; t < 1024; t += 256) {
      int i = t >> 4, seg = t & 15;
      async_copy16(&XCb[(tok0 + R * 64 + i) * CONVDIM + (DINNER + DSTATE) + seg * 8],
                   &Cr[i][seg * 8]);
    }
    if (tid < 64) acR[tid] = ACUM[acb + R * 64 + tid];
    async_wait();
    __syncthreads();

    v8f accY[2];
    accY[0] = zero8(); accY[1] = zero8();

    for (int S = 0; S <= R; ++S) {
      for (int t = tid; t < 1024; t += 256) {
        int j = t >> 4, seg = t & 15;
        async_copy16(&XCb[(tok0 + S * 64 + j) * CONVDIM + DINNER + seg * 8],
                     &Bsl[j][seg * 8]);
      }
      // x^T panel: pure bf16 transpose copy (dt is folded into G below)
      for (int t = tid; t < 64 * 64; t += 256) {
        int p = t & 63, j = t >> 6;
        XT[p][j] = XCb[(tok0 + S * 64 + j) * CONVDIM + h * HEADDIM + p];
      }
      if (tid < 64) {
        acS[tid] = ACUM[acb + S * 64 + tid];
        dtS[tid] = DT[(tok0 + S * 64 + tid) * NHEADS + h];
      }
      async_wait();
      __syncthreads();

      // G = C_R @ B_S^T (K=128), mask with decay * dt, store bf16 to LDS
      #pragma unroll
      for (int t2 = 0; t2 < 2; ++t2) {
        int j0 = nb + t2 * 16;
        v8f g = zero8();
        #pragma unroll
        for (int kk = 0; kk < 4; ++kk) {
          v16bf a  = frag_a(&Cr[0][0], m0, 128, kk * 32, lane);
          v16bf bf = frag_b(&Bsl[0][0], j0, 128, kk * 32, lane);
          g = wmma_bf16(a, bf, g);
        }
        #pragma unroll
        for (int r = 0; r < 8; ++r) {
          int i = m0 + r + row_hi;
          int j = j0 + colL;
          int gi = R * 64 + i, gj = S * 64 + j;
          float mlt = (gj <= gi) ? __expf(acR[i] - acS[j]) * dtS[j] : 0.f;
          Gs[i][j] = f32_bf16r(g[r] * mlt);
        }
      }
      __syncthreads();

      // Y += G @ x_S, K = 64
      #pragma unroll
      for (int t2 = 0; t2 < 2; ++t2) {
        int p0 = nb + t2 * 16;
        #pragma unroll
        for (int kk = 0; kk < 2; ++kk) {
          v16bf a  = frag_a(&Gs[0][0], m0, 64, kk * 32, lane);
          v16bf bf = frag_b(&XT[0][0], p0, 64, kk * 32, lane);
          accY[t2] = wmma_bf16(a, bf, accY[t2]);
        }
      }
      __syncthreads();
    }

    // store Y_diag block
    #pragma unroll
    for (int t2 = 0; t2 < 2; ++t2) {
      #pragma unroll
      for (int r = 0; r < 8; ++r) {
        long gl = tok0 + R * 64 + m0 + r + row_hi;
        int  p  = nb + t2 * 16 + colL;
        Y[gl * DINNER + h * HEADDIM + p] = accY[t2][r];
      }
    }
    __syncthreads();
  }

  // ---------- phase 2: states[n,p] = sum_l B[l,n]*exp(ac_last-ac_l)*dt_l*x[l,p] ---
  unsigned short* BdT = &Cr[0][0];           // reuse as [n][l] : 128x64
  const float aclast = ACUM[acb + CHUNK - 1];
  const int sn0 = (wave >> 1) * 32;          // n band (32 of 128)
  const int sp0 = (wave & 1) * 32;           // p band (32 of 64)
  v8f accS_[4];
  #pragma unroll
  for (int q = 0; q < 4; ++q) accS_[q] = zero8();

  for (int S = 0; S < 4; ++S) {
    if (tid < 64) {
      acS[tid] = ACUM[acb + S * 64 + tid];
      dtS[tid] = DT[(tok0 + S * 64 + tid) * NHEADS + h];
    }
    __syncthreads();
    for (int t = tid; t < 128 * 64; t += 256) {
      int n = t & 127, j = t >> 7;
      float bv = bf16_f32(XCb[(tok0 + S * 64 + j) * CONVDIM + DINNER + n]);
      float sc = __expf(aclast - acS[j]) * dtS[j];
      BdT[n * 64 + j] = f32_bf16r(bv * sc);
    }
    for (int t = tid; t < 64 * 64; t += 256) {
      int p = t & 63, j = t >> 6;
      XT[p][j] = XCb[(tok0 + S * 64 + j) * CONVDIM + h * HEADDIM + p];
    }
    __syncthreads();
    #pragma unroll
    for (int tn = 0; tn < 2; ++tn)
      #pragma unroll
      for (int tp = 0; tp < 2; ++tp) {
        int n0 = sn0 + tn * 16, p0 = sp0 + tp * 16;
        #pragma unroll
        for (int kk = 0; kk < 2; ++kk) {
          v16bf a  = frag_a(BdT, n0, 64, kk * 32, lane);
          v16bf bf = frag_b(&XT[0][0], p0, 64, kk * 32, lane);
          accS_[tn * 2 + tp] = wmma_bf16(a, bf, accS_[tn * 2 + tp]);
        }
      }
    __syncthreads();
  }
  // store to (p, n) layout: STATES[bid][p][n]
  #pragma unroll
  for (int tn = 0; tn < 2; ++tn)
    #pragma unroll
    for (int tp = 0; tp < 2; ++tp) {
      #pragma unroll
      for (int r = 0; r < 8; ++r) {
        int n = sn0 + tn * 16 + r + row_hi;
        int p = sp0 + tp * 16 + colL;
        STATES[(long)bid * (HEADDIM * DSTATE) + p * DSTATE + n] = accS_[tn * 2 + tp][r];
      }
    }
}

// ================= K5: inter-chunk sequential scan (bf16 prev out) =================
__global__ __launch_bounds__(256) void k_scan(const float* __restrict__ STATES,
                                              const float* __restrict__ ACUM,
                                              unsigned short* __restrict__ PREVb) {
  long idx = (long)blockIdx.x * 256 + threadIdx.x;   // over B*H*P*N = 524288
  if (idx >= (long)NBATCH * NHEADS * HEADDIM * DSTATE) return;
  int n = (int)(idx & 127);
  int p = (int)((idx >> 7) & 63);
  int h = (int)((idx >> 13) & 31);
  int b = (int)(idx >> 18);
  float carry = 0.f;
  for (int c = 0; c < NCHUNK; ++c) {
    int  bid  = (b * NCHUNK + c) * NHEADS + h;
    long soff = (long)bid * (HEADDIM * DSTATE) + p * DSTATE + n;
    PREVb[soff] = f32_bf16r(carry);
    float d = __expf(ACUM[(long)bid * CHUNK + CHUNK - 1]);
    carry = carry * d + STATES[soff];
  }
}

// ================= K6: Y_off = exp(acum)*(C @ prev^T); Y += Y_off + D*x ==========
__global__ __launch_bounds__(256) void k_yoff(const unsigned short* __restrict__ XCb,
                                              const unsigned short* __restrict__ PREVb,
                                              const float* __restrict__ ACUM,
                                              const float* __restrict__ D_param,
                                              float* __restrict__ Y) {
  __shared__ __align__(16) unsigned short Pt[64][128];   // prev [p][n]
  __shared__ __align__(16) unsigned short Cs[64][128];   // C rows [i][n]
  __shared__ float acl[64];
  const int bid  = blockIdx.x;
  const int h    = bid & 31;
  const int bc   = bid >> 5;
  const int b    = bc >> 3, c = bc & 7;
  const int tid  = threadIdx.x;
  const int lane = tid & 31;
  const int wave = tid >> 5;
  const int row_hi = (lane >> 4) ? 8 : 0;
  const int colL   = lane & 15;
  const long tok0  = (long)b * SEQLEN + c * CHUNK;
  const long acb   = (long)bid * CHUNK;

  // async-stage prev[p][n] (contiguous 16KB)
  for (int t = tid; t < 1024; t += 256)
    async_copy16(&PREVb[(long)bid * (HEADDIM * DSTATE) + t * 8], &Pt[0][0] + t * 8);
  const float Dh = D_param[h];
  const int m0 = (wave >> 1) * 16;
  const int pb = (wave & 1) * 32;

  for (int R = 0; R < 4; ++R) {
    for (int t = tid; t < 1024; t += 256) {
      int i = t >> 4, seg = t & 15;
      async_copy16(&XCb[(tok0 + R * 64 + i) * CONVDIM + (DINNER + DSTATE) + seg * 8],
                   &Cs[i][seg * 8]);
    }
    if (tid < 64) acl[tid] = ACUM[acb + R * 64 + tid];
    async_wait();
    __syncthreads();
    #pragma unroll
    for (int t2 = 0; t2 < 2; ++t2) {
      int p0 = pb + t2 * 16;
      v8f acc = zero8();
      #pragma unroll
      for (int kk = 0; kk < 4; ++kk) {
        v16bf a  = frag_a(&Cs[0][0], m0, 128, kk * 32, lane);
        v16bf bf = frag_b(&Pt[0][0], p0, 128, kk * 32, lane);
        acc = wmma_bf16(a, bf, acc);
      }
      #pragma unroll
      for (int r = 0; r < 8; ++r) {
        int i = m0 + r + row_hi;
        int p = p0 + colL;
        long gl = tok0 + R * 64 + i;
        float xv = bf16_f32(XCb[gl * CONVDIM + h * HEADDIM + p]);
        long yo = gl * DINNER + h * HEADDIM + p;
        Y[yo] = Y[yo] + __expf(acl[i]) * acc[r] + Dh * xv;
      }
    }
    __syncthreads();
  }
}

// ================= K7: gated RMSNorm =================
__global__ __launch_bounds__(256) void k_norm(const float* __restrict__ Y,
                                              const float* __restrict__ ZX,
                                              const float* __restrict__ nw,
                                              float* __restrict__ OUT) {
  __shared__ float red[8];
  long t = blockIdx.x;                        // token
  int tid = threadIdx.x, lane = tid & 31, wave = tid >> 5;
  const float* y = Y + t * DINNER;
  const float* z = ZX + t * DPROJ;            // z = first 2048 cols of zxbcdt
  float g[8];
  float ss = 0.f;
  #pragma unroll
  for (int i = 0; i < 8; ++i) {
    int cidx = tid + i * 256;
    float gv = y[cidx] * silu(z[cidx]);
    g[i] = gv;
    ss += gv * gv;
  }
  #pragma unroll
  for (int off = 16; off > 0; off >>= 1) ss += __shfl_down(ss, off, 32);
  if (lane == 0) red[wave] = ss;
  __syncthreads();
  if (tid == 0) {
    float s2 = 0.f;
    for (int w = 0; w < 8; ++w) s2 += red[w];
    red[0] = rsqrtf(s2 / (float)DINNER + 1e-5f);
  }
  __syncthreads();
  float scale = red[0];
  #pragma unroll
  for (int i = 0; i < 8; ++i) {
    int cidx = tid + i * 256;
    OUT[t * DINNER + cidx] = g[i] * scale * nw[cidx];
  }
}

// ================= launch =================
extern "C" void kernel_launch(void* const* d_in, const int* in_sizes, int n_in,
                              void* d_out, int out_size, void* d_ws, size_t ws_size,
                              hipStream_t stream) {
  const float* U       = (const float*)d_in[0];
  const float* W_in    = (const float*)d_in[1];
  const float* conv_w  = (const float*)d_in[2];
  const float* conv_b  = (const float*)d_in[3];
  const float* dt_bias = (const float*)d_in[4];
  const float* A_log   = (const float*)d_in[5];
  const float* D_param = (const float*)d_in[6];
  const float* norm_w  = (const float*)d_in[7];
  float* OUT = (float*)d_out;

  char* wsp = (char*)d_ws;
  float* ZX = (float*)wsp;                 wsp += (size_t)BL * DPROJ * 4;
  unsigned short* Ub  = (unsigned short*)wsp; wsp += (size_t)BL * DMODEL * 2;
  unsigned short* Wb  = (unsigned short*)wsp; wsp += (size_t)DPROJ * DMODEL * 2;
  unsigned short* XCb = (unsigned short*)wsp; wsp += (size_t)BL * CONVDIM * 2;
  float* DT     = (float*)wsp;             wsp += (size_t)BL * NHEADS * 4;
  float* ACUM   = (float*)wsp;             wsp += (size_t)NBCH * CHUNK * 4;
  float* STATES = (float*)wsp;             wsp += (size_t)NBCH * HEADDIM * DSTATE * 4;
  unsigned short* PREVb = (unsigned short*)wsp; wsp += (size_t)NBCH * HEADDIM * DSTATE * 2;
  float* Ybuf   = (float*)wsp;

  const long nU = (long)BL * DMODEL;
  const long nW = (long)DPROJ * DMODEL;
  k_cvt<<<(int)((nU + 255) / 256), 256, 0, stream>>>(U, Ub, nU);
  k_cvt<<<(int)((nW + 255) / 256), 256, 0, stream>>>(W_in, Wb, nW);
  k_in_proj<<<dim3((DPROJ + BLK_N - 1) / BLK_N, BL / BLK_M), 256, 0, stream>>>(Ub, Wb, ZX);
  k_conv<<<(int)(((long)BL * CONVDIM + 255) / 256), 256, 0, stream>>>(ZX, conv_w, conv_b, XCb);
  k_dt<<<(BL * NHEADS + 255) / 256, 256, 0, stream>>>(ZX, dt_bias, DT);
  k_cumsum<<<NBCH, 256, 0, stream>>>(DT, A_log, ACUM);
  k_chunk<<<NBCH, 256, 0, stream>>>(XCb, DT, ACUM, Ybuf, STATES);
  k_scan<<<(NBATCH * NHEADS * HEADDIM * DSTATE + 255) / 256, 256, 0, stream>>>(STATES, ACUM, PREVb);
  k_yoff<<<NBCH, 256, 0, stream>>>(XCb, PREVb, ACUM, D_param, Ybuf);
  k_norm<<<BL, 256, 0, stream>>>(Ybuf, ZX, norm_w, OUT);
}
